// GPSE_45286135169328
// MI455X (gfx1250) — compile-verified, matrix operand order checked by reading, not run on hardware
//
#include <hip/hip_runtime.h>
#include <hip/hip_bf16.h>

#define N_NODES 50000
#define N_EDGES 800000
#define DIM_IN 20
#define DIM 128
#define NCAT 512            // [k | q | v | agg] concatenated per row
#define BN_EPS 1e-5f
#define L2_EPS 1e-12f

typedef __attribute__((ext_vector_type(16))) __bf16 v16bf;
typedef __attribute__((ext_vector_type(8)))  float  v8f;
typedef __attribute__((ext_vector_type(4)))  unsigned int v4u;
typedef __attribute__((ext_vector_type(8)))  int    v8i;
typedef __attribute__((ext_vector_type(4)))  int    v4i;
typedef unsigned short u16;

static __device__ __forceinline__ u16 f2bf(float f) {
    __hip_bfloat16 h = __float2bfloat16(f);
    return *reinterpret_cast<u16*>(&h);
}

// ---------------------------------------------------------------------------
// Tensor Data Mover: async copy of `n8*8` contiguous bytes from global to LDS.
// D# per ISA ch.8: group0 = {count=1, lds_addr, global_addr[56:0], type=2},
// group1 = {data_size=8B, tensor_dim0=tile_dim0=n8, tensor_dim1=tile_dim1=1,
//           dim0_stride=n8}. Issued once per wave; tracked with TENSORcnt.
// ---------------------------------------------------------------------------
static __device__ __forceinline__ void tdm_load_1d(unsigned lds_addr,
                                                   const void* gptr,
                                                   unsigned n8) {
    unsigned long long ga = (unsigned long long)(size_t)gptr;
    v4u g0;
    g0[0] = 1u;                                               // count=1 (valid)
    g0[1] = lds_addr;                                         // LDS byte address
    g0[2] = (unsigned)ga;                                     // global addr lo
    g0[3] = ((unsigned)(ga >> 32) & 0x01FFFFFFu) | 0x80000000u; // hi + type=2
    v8i g1;
    g1[0] = (int)(3u << 16);                 // wg_mask=0, data_size=3 (8 bytes)
    g1[1] = (int)((n8 & 0xFFFFu) << 16);     // tensor_dim0 lo16 @ bits[63:48]
    g1[2] = (int)(((n8 >> 16) & 0xFFFFu) | (1u << 16)); // dim0 hi16, tensor_dim1=1
    g1[3] = (int)(n8 << 16);                 // tile_dim0 = n8 (<=65535)
    g1[4] = 1;                               // tile_dim1 = 1, tile_dim2 = 0
    g1[5] = (int)n8;                         // tensor_dim0_stride lo32
    g1[6] = 0;
    g1[7] = 0;
    v4i gz = {0, 0, 0, 0};
#if defined(__clang_major__) && (__clang_major__ >= 23)
    v8i gz8 = {0, 0, 0, 0, 0, 0, 0, 0};
    __builtin_amdgcn_tensor_load_to_lds(g0, g1, gz, gz, gz8, 0);
#else
    __builtin_amdgcn_tensor_load_to_lds(g0, g1, gz, gz, 0);
#endif
}

// ---------------------------------------------------------------------------
// Convert + transpose the 4 per-layer weight matrices into one bf16
// column-major [512 x 128] B matrix per layer: wcat[l][n_global*128 + k].
// n_global: 0-127 = wk, 128-255 = wq, 256-383 = wv, 384-511 = ws.
// ---------------------------------------------------------------------------
__global__ void convert_weights(const float* __restrict__ wk,
                                const float* __restrict__ wq,
                                const float* __restrict__ wv,
                                const float* __restrict__ wsk,
                                u16* __restrict__ wcat) {
    int idx = blockIdx.x * 256 + threadIdx.x;      // 0 .. 2*512*128-1
    int l   = idx >> 16;
    int rem = idx & 65535;
    int ng  = rem >> 7;                            // n_global 0..511
    int k   = rem & 127;
    int g   = ng >> 7;
    int n   = ng & 127;
    const float* w = (g == 0) ? wk : (g == 1) ? wq : (g == 2) ? wv : wsk;
    wcat[idx] = f2bf(w[l * DIM * DIM + k * DIM + n]);
}

// ---------------------------------------------------------------------------
// Pre-MP linear: x[50000x20] @ pre_w[20x128] -> out[50000x128] (f32).
// ---------------------------------------------------------------------------
__global__ void gemm_pre(const float* __restrict__ x,
                         const float* __restrict__ w,
                         float* __restrict__ out) {
    __shared__ float wl[DIM_IN * DIM];
    int t = threadIdx.x;
    for (int i = t; i < DIM_IN * DIM; i += 256) wl[i] = w[i];
    __syncthreads();
    int row = blockIdx.x * 2 + (t >> 7);
    int col = t & 127;
    if (row >= N_NODES) return;
    const float* xr = x + row * DIM_IN;
    float acc = 0.f;
#pragma unroll
    for (int i = 0; i < DIM_IN; i++) acc += xr[i] * wl[i * DIM + col];
    out[(size_t)row * DIM + col] = acc;
}

// ---------------------------------------------------------------------------
// BatchNorm stats over the node axis, folded into y = a*x + b.
// ---------------------------------------------------------------------------
__global__ void bn_stats(const float* __restrict__ in, int stride,
                         const float* __restrict__ gamma,
                         const float* __restrict__ beta,
                         float* __restrict__ ab) {
    int c = blockIdx.x;
    int t = threadIdx.x;
    float s = 0.f, s2 = 0.f;
    for (int r = t; r < N_NODES; r += 256) {
        float v = in[(size_t)r * stride + c];
        s += v;
        s2 += v * v;
    }
    __shared__ float sh0[256];
    __shared__ float sh1[256];
    sh0[t] = s; sh1[t] = s2;
    __syncthreads();
    for (int o = 128; o > 0; o >>= 1) {
        if (t < o) { sh0[t] += sh0[t + o]; sh1[t] += sh1[t + o]; }
        __syncthreads();
    }
    if (t == 0) {
        float mean = sh0[0] * (1.0f / N_NODES);
        float var  = sh1[0] * (1.0f / N_NODES) - mean * mean;
        float a = gamma[c] * rsqrtf(var + BN_EPS);
        ab[c]       = a;
        ab[DIM + c] = beta[c] - mean * a;
    }
}

// ---------------------------------------------------------------------------
// Fused BN apply + ReLU + row l2-normalize. One wave32 per node row.
// ---------------------------------------------------------------------------
__global__ void bn_relu_l2norm(const float* __restrict__ in, int stride,
                               const float* __restrict__ ab,
                               u16* __restrict__ hbf,
                               float* __restrict__ fout) {
    int wid  = (blockIdx.x * blockDim.x + threadIdx.x) >> 5;
    int lane = threadIdx.x & 31;
    if (wid >= N_NODES) return;
    const float* r = in + (size_t)wid * stride + lane * 4;
    float4 xv = *(const float4*)r;
    float4 av = *(const float4*)(ab + lane * 4);
    float4 bv = *(const float4*)(ab + DIM + lane * 4);
    float y0 = fmaxf(xv.x * av.x + bv.x, 0.f);
    float y1 = fmaxf(xv.y * av.y + bv.y, 0.f);
    float y2 = fmaxf(xv.z * av.z + bv.z, 0.f);
    float y3 = fmaxf(xv.w * av.w + bv.w, 0.f);
    float ss = y0 * y0 + y1 * y1 + y2 * y2 + y3 * y3;
#pragma unroll
    for (int m = 16; m > 0; m >>= 1) ss += __shfl_xor(ss, m, 32);
    float sc = 1.f / fmaxf(sqrtf(ss), L2_EPS);
    y0 *= sc; y1 *= sc; y2 *= sc; y3 *= sc;
    ushort4 p;
    p.x = f2bf(y0); p.y = f2bf(y1); p.z = f2bf(y2); p.w = f2bf(y3);
    *(ushort4*)(hbf + (size_t)wid * DIM + lane * 4) = p;
    if (fout) {
        *(float4*)(fout + (size_t)wid * DIM + lane * 4) = make_float4(y0, y1, y2, y3);
    }
}

// ---------------------------------------------------------------------------
// Fused 4-way GEMM: hbf[50000x128] (bf16) @ wcat[128x512] (bf16, col-major)
// -> kqvs[50000x512] (f32), via v_wmma_f32_16x16x32_bf16.
//
// Per block (8 waves): TDM stages the 128KB B panel + this block's 4KB A tile
// into LDS (one async tensor_load_to_lds each, issued by wave 0, waited with
// s_wait_tensorcnt). Inner loop then feeds WMMA purely from ds_load_b128 —
// no VMEM in the MAC loop. Each wave produces a 16x64 strip (4 acc tiles).
//
// 16-bit A/B per-lane layout (ISA 7.12.2):
//   lanes 0-15  : row=lane,    K in {k0+0..7,  k0+16..23}
//   lanes 16-31 : row=lane-16, K in {k0+8..15, k0+24..31}
// -> two 16B LDS loads per operand per k-step.
// ---------------------------------------------------------------------------
__global__ void __launch_bounds__(256) gemm_wmma(const u16* __restrict__ hbf,
                                                 const u16* __restrict__ wcat,
                                                 float* __restrict__ out) {
    __shared__ u16 sB[NCAT * DIM];   // 128 KB weight panel, col-major
    __shared__ u16 sA[16 * DIM];     // 4 KB row tile

    int wave = threadIdx.x >> 5;     // 0..7 -> 64-column group
    int lane = threadIdx.x & 31;
    int rt   = blockIdx.x;           // row tile

    if (wave == 0) {
        tdm_load_1d((unsigned)(size_t)&sB[0], wcat, (NCAT * DIM * 2) / 8);
        tdm_load_1d((unsigned)(size_t)&sA[0], hbf + (size_t)rt * 16 * DIM,
                    (16 * DIM * 2) / 8);
        __builtin_amdgcn_s_wait_tensorcnt(0);
    }
    __syncthreads();

    int m    = lane & 15;
    int hsel = (lane >> 4) * 8;      // 0 or 8 halves
    int nb0  = wave * 64;

    v8f acc[4] = {v8f{}, v8f{}, v8f{}, v8f{}};

#pragma unroll
    for (int ks = 0; ks < 4; ks++) {
        int k0 = ks * 32;
        union { uint4 u[2]; v16bf v; } A;
        const u16* ar = &sA[m * DIM];
        A.u[0] = *(const uint4*)(ar + k0 + hsel);
        A.u[1] = *(const uint4*)(ar + k0 + hsel + 16);
        union { uint4 u[2]; v16bf v; } B[4];
#pragma unroll
        for (int t = 0; t < 4; t++) {
            const u16* bc = &sB[(nb0 + t * 16 + m) * DIM];
            B[t].u[0] = *(const uint4*)(bc + k0 + hsel);
            B[t].u[1] = *(const uint4*)(bc + k0 + hsel + 16);
        }
#pragma unroll
        for (int t = 0; t < 4; t++) {
            acc[t] = __builtin_amdgcn_wmma_f32_16x16x32_bf16(
                false, A.v, false, B[t].v, (short)0, acc[t], false, false);
        }
    }

    // C/D layout: lanes 0-15 -> N=lane, VGPR r = row M=r; lanes 16-31 -> M=8+r
    int rbase = rt * 16 + (lane >> 4) * 8;
    int cbase = nb0 + (lane & 15);
#pragma unroll
    for (int t = 0; t < 4; t++) {
#pragma unroll
        for (int r = 0; r < 8; r++) {
            out[(size_t)(rbase + r) * NCAT + cbase + t * 16] = acc[t][r];
        }
    }
}

// ---------------------------------------------------------------------------
// Edge phase: one wave32 per edge, 4 dims/lane.
//   msg = sigmoid(k[dst] + q[src]) * v[src];  agg[dst] += msg (f32 atomics).
// ---------------------------------------------------------------------------
__global__ void edge_msg(const int* __restrict__ ei, float* __restrict__ kqvs) {
    int wid  = (blockIdx.x * blockDim.x + threadIdx.x) >> 5;
    int lane = threadIdx.x & 31;
    if (wid >= N_EDGES) return;
    int s = ei[wid];             // src
    int d = ei[N_EDGES + wid];   // dst
    const float4 kd = *(const float4*)(kqvs + (size_t)d * NCAT + 0   + lane * 4);
    const float4 qs = *(const float4*)(kqvs + (size_t)s * NCAT + 128 + lane * 4);
    const float4 vs = *(const float4*)(kqvs + (size_t)s * NCAT + 256 + lane * 4);
    float g0 = 1.f / (1.f + __expf(-(kd.x + qs.x)));
    float g1 = 1.f / (1.f + __expf(-(kd.y + qs.y)));
    float g2 = 1.f / (1.f + __expf(-(kd.z + qs.z)));
    float g3 = 1.f / (1.f + __expf(-(kd.w + qs.w)));
    float* agg = kqvs + (size_t)d * NCAT + 384 + lane * 4;
    unsafeAtomicAdd(agg + 0, g0 * vs.x);
    unsafeAtomicAdd(agg + 1, g1 * vs.y);
    unsafeAtomicAdd(agg + 2, g2 * vs.z);
    unsafeAtomicAdd(agg + 3, g3 * vs.w);
}

// ---------------------------------------------------------------------------
// Orchestration
// ---------------------------------------------------------------------------
extern "C" void kernel_launch(void* const* d_in, const int* in_sizes, int n_in,
                              void* d_out, int out_size, void* d_ws, size_t ws_size,
                              hipStream_t stream) {
    const float* x         = (const float*)d_in[0];
    const int*   ei        = (const int*)d_in[1];
    const float* pre_w     = (const float*)d_in[2];
    const float* pre_gamma = (const float*)d_in[3];
    const float* pre_beta  = (const float*)d_in[4];
    const float* mp_wk     = (const float*)d_in[5];
    const float* mp_wq     = (const float*)d_in[6];
    const float* mp_wv     = (const float*)d_in[7];
    const float* mp_ws     = (const float*)d_in[8];
    const float* mp_gamma  = (const float*)d_in[9];
    const float* mp_beta   = (const float*)d_in[10];

    char* ws = (char*)d_ws;
    // layout: hbf (bf16 features) | kqvs (f32 k|q|v|agg) | ab coeffs | bf16 weights
    u16*   hbf  = (u16*)ws;                                    // 12.8 MB
    float* kqvs = (float*)(ws + ((size_t)16 << 20));           // 102.4 MB
    float* ab   = (float*)(ws + ((size_t)120 << 20));          // 1 KB
    u16*   wcat = (u16*)(ws + ((size_t)120 << 20) + 4096);     // 256 KB

    // 0) weights -> bf16, column-major, concatenated per layer
    convert_weights<<<512, 256, 0, stream>>>(mp_wk, mp_wq, mp_wv, mp_ws, wcat);

    // 1) pre-MP: linear -> BN -> ReLU -> l2norm (tmp in kqvs, stride DIM)
    gemm_pre<<<25000, 256, 0, stream>>>(x, pre_w, kqvs);
    bn_stats<<<128, 256, 0, stream>>>(kqvs, DIM, pre_gamma, pre_beta, ab);
    bn_relu_l2norm<<<6250, 256, 0, stream>>>(kqvs, DIM, ab, hbf, nullptr);

    // 2) message-passing layers
    for (int l = 0; l < 2; l++) {
        gemm_wmma<<<3125, 256, 0, stream>>>(hbf, wcat + (size_t)l * 512 * DIM, kqvs);
        edge_msg<<<100000, 256, 0, stream>>>(ei, kqvs);
        bn_stats<<<128, 256, 0, stream>>>(kqvs + 384, NCAT,
                                          mp_gamma + l * DIM, mp_beta + l * DIM, ab);
        bn_relu_l2norm<<<6250, 256, 0, stream>>>(kqvs + 384, NCAT, ab, hbf,
                                                 (l == 1) ? (float*)d_out : nullptr);
    }
}
